// TriplaneDispFields_3298534884024
// MI455X (gfx1250) — matrix operand order repaired and checked by reading, not compiled, not run on hardware
//
#include <hip/hip_runtime.h>
#include <hip/hip_bf16.h>

typedef __attribute__((ext_vector_type(16))) __bf16          v16bf;
typedef __attribute__((ext_vector_type(8)))  float           v8f;
typedef __attribute__((ext_vector_type(8)))  unsigned short  us8v;

#define WMMA_BF16(A, B, C) \
  __builtin_amdgcn_wmma_f32_16x16x32_bf16(false, (A), false, (B), (short)0, (C), false, false)

// ---------------- constants / LDS layout (units: unsigned short) ----------------
#define WAVES        8
#define BLOCK        256
#define ROW_STRIDE   264   // 16 act rows * 528B; 528B = 132 dwords -> rows offset by 4 banks

#define W0_OFF   0        // dW0  64x192
#define W1_OFF   12288    // dW1  64x64
#define W2_OFF   16384    // dW2  64x256
#define W3_OFF   32768    // dW3  64x64
#define W4_OFF   36864    // dW4  64x256
#define W5_OFF   53248    // dW5  padded 16x64
#define C0_OFF   54272    // cW0  64x64
#define C1_OFF   58368    // cW1  padded 16x64
#define WEIGHT_US 59392

// biases (float), after weights: b0,b1,b2,b3,b4 (64 each), b5(16 pad), cb0(64), cb1(16 pad)
#define NBIAS    416
#define B0_F 0
#define B1_F 64
#define B2_F 128
#define B3_F 192
#define B4_F 256
#define B5_F 320
#define CB0_F 336
#define CB1_F 400

#define ACT_OFF      (WEIGHT_US + 2*NBIAS)        // 60224
#define ACT_PER_WAVE (16*ROW_STRIDE)              // 4224
#define DISP_OFF_US  (ACT_OFF + WAVES*ACT_PER_WAVE) // 94016
#define TOTAL_US     (DISP_OFF_US + WAVES*64)     // 94528 ushort = 189056 B
#define SHMEM_BYTES  (TOTAL_US*2)

// ---------------- helpers ----------------
__device__ __forceinline__ unsigned short f2bf(float f) {
  unsigned int u = __float_as_uint(f);
  u += 0x7FFFu + ((u >> 16) & 1u);          // round-to-nearest-even
  return (unsigned short)(u >> 16);
}

// A fragment: 16x32 bf16, lane = half*16+m holds row m, K = kBase + half*16 + j
__device__ __forceinline__ v16bf loadA(const unsigned short* act, int m, int half, int kBase) {
  union { v16bf v; us8v h[2]; } u;
  const unsigned short* p = act + m * ROW_STRIDE + kBase + half * 16;
  u.h[0] = *(const us8v*)(p);
  u.h[1] = *(const us8v*)(p + 8);
  return u.v;
}

// B fragment: 32x16 bf16 (W stored row-major [n][k]); lane holds col n=m, K = kt + half*16 + j
__device__ __forceinline__ v16bf loadB(const unsigned short* w, int ldk, int n, int half,
                                       int kt, int nt) {
  union { v16bf v; us8v h[2]; } u;
  const unsigned short* p = w + (nt * 16 + n) * ldk + kt + half * 16;
  u.h[0] = *(const us8v*)(p);
  u.h[1] = *(const us8v*)(p + 8);
  return u.v;
}

// 16x16 output tile (NT=1), f32 acc
__device__ __forceinline__ v8f gemm16(const unsigned short* actIn, int kBase, int K,
                                      const unsigned short* w, int ldk, int m, int half) {
  v8f acc = {};
#pragma unroll
  for (int kt = 0; kt < 64; kt += 32) {           // K==64 for both users
    (void)K;
    v16bf a = loadA(actIn, m, half, kBase + kt);
    v16bf b = loadB(w, ldk, m, half, kt, 0);
    acc = WMMA_BF16(a, b, acc);
  }
  return acc;
}

// hidden layer: [16 x K] @ [K x 64] + bias, ReLU, bf16 back to LDS at outCol
template <int K>
__device__ __forceinline__ void mlp_layer64(const unsigned short* actIn, int kBase,
                                            const unsigned short* w, int ldk,
                                            const float* bias,
                                            unsigned short* actOut, int outCol,
                                            int m, int half) {
  v8f z = {};
  v8f acc0 = z, acc1 = z, acc2 = z, acc3 = z;
#pragma unroll
  for (int kt = 0; kt < K; kt += 32) {
    v16bf a = loadA(actIn, m, half, kBase + kt);
    acc0 = WMMA_BF16(a, loadB(w, ldk, m, half, kt, 0), acc0);
    acc1 = WMMA_BF16(a, loadB(w, ldk, m, half, kt, 1), acc1);
    acc2 = WMMA_BF16(a, loadB(w, ldk, m, half, kt, 2), acc2);
    acc3 = WMMA_BF16(a, loadB(w, ldk, m, half, kt, 3), acc3);
  }
  const int rb = half * 8;
  const float b0 = bias[m], b1 = bias[16 + m], b2 = bias[32 + m], b3 = bias[48 + m];
#pragma unroll
  for (int r = 0; r < 8; ++r) {
    unsigned short* row = actOut + (rb + r) * ROW_STRIDE + outCol;
    row[m]      = f2bf(fmaxf(acc0[r] + b0, 0.f));
    row[16 + m] = f2bf(fmaxf(acc1[r] + b1, 0.f));
    row[32 + m] = f2bf(fmaxf(acc2[r] + b2, 0.f));
    row[48 + m] = f2bf(fmaxf(acc3[r] + b3, 0.f));
  }
}

// bilinear grid_sample (align_corners=True, border clamp); each lane does 32 channels
__device__ __forceinline__ void sample_plane(const float* plane, int W, int H,
                                             float gx, float gy,
                                             unsigned short* act, int m, int half, int colBase) {
  float fx = fminf(fmaxf((gx + 1.f) * 0.5f * (float)(W - 1), 0.f), (float)(W - 1));
  float fy = fminf(fmaxf((gy + 1.f) * 0.5f * (float)(H - 1), 0.f), (float)(H - 1));
  int x0 = (int)fx, y0 = (int)fy;
  int x1 = (x0 + 1 < W) ? x0 + 1 : W - 1;
  int y1 = (y0 + 1 < H) ? y0 + 1 : H - 1;
  float wx = fx - (float)x0, wy = fy - (float)y0;
  float w00 = (1.f - wx) * (1.f - wy), w01 = wx * (1.f - wy);
  float w10 = (1.f - wx) * wy,         w11 = wx * wy;
  int i00 = y0 * W + x0, i01 = y0 * W + x1, i10 = y1 * W + x0, i11 = y1 * W + x1;
  const int HW = W * H;
  const float* b = plane + (half * 32) * HW;
  unsigned short* dst = act + m * ROW_STRIDE + colBase + half * 32;
#pragma unroll 4
  for (int c = 0; c < 32; ++c) {
    float v = b[i00] * w00 + b[i01] * w01 + b[i10] * w10 + b[i11] * w11;
    dst[c] = f2bf(v);
    b += HW;
  }
}

__device__ __forceinline__ void stage_w(const float* src, int rows, int padRows, int cols,
                                        unsigned short* dst, int tid) {
  const int total = padRows * cols;
  for (int i = tid; i < total; i += BLOCK) {
    int r = i / cols, c = i - r * cols;
    dst[i] = f2bf(r < rows ? src[r * cols + c] : 0.f);
  }
}

// ---------------- kernel ----------------
extern __shared__ unsigned short lds_us[];

__global__ void __launch_bounds__(BLOCK, 1)
triplane_disp_kernel(const float* __restrict__ inp, const float* __restrict__ aabb,
                     const float* __restrict__ plane01, const float* __restrict__ plane02,
                     const float* __restrict__ plane12, const float* __restrict__ can_plane,
                     const float* __restrict__ dW0, const float* __restrict__ db0,
                     const float* __restrict__ dW1, const float* __restrict__ db1,
                     const float* __restrict__ dW2, const float* __restrict__ db2,
                     const float* __restrict__ dW3, const float* __restrict__ db3,
                     const float* __restrict__ dW4, const float* __restrict__ db4,
                     const float* __restrict__ dW5, const float* __restrict__ db5,
                     const float* __restrict__ cW0, const float* __restrict__ cb0,
                     const float* __restrict__ cW1, const float* __restrict__ cb1,
                     float* __restrict__ out, int nPoints) {
  const int tid  = threadIdx.x;
  const int wave = tid >> 5;
  const int lane = tid & 31;
  const int m    = lane & 15;
  const int half = lane >> 4;

  // ---- stage weights (f32 -> bf16) and biases into LDS, once per block ----
  stage_w(dW0, 64, 64, 192, lds_us + W0_OFF, tid);
  stage_w(dW1, 64, 64,  64, lds_us + W1_OFF, tid);
  stage_w(dW2, 64, 64, 256, lds_us + W2_OFF, tid);
  stage_w(dW3, 64, 64,  64, lds_us + W3_OFF, tid);
  stage_w(dW4, 64, 64, 256, lds_us + W4_OFF, tid);
  stage_w(dW5,  2, 16,  64, lds_us + W5_OFF, tid);
  stage_w(cW0, 64, 64,  64, lds_us + C0_OFF, tid);
  stage_w(cW1,  3, 16,  64, lds_us + C1_OFF, tid);

  float* biasF = (float*)(lds_us + WEIGHT_US);
  for (int i = tid; i < NBIAS; i += BLOCK) {
    float v;
    if      (i < 64)  v = db0[i];
    else if (i < 128) v = db1[i - 64];
    else if (i < 192) v = db2[i - 128];
    else if (i < 256) v = db3[i - 192];
    else if (i < 320) v = db4[i - 256];
    else if (i < 336) v = (i - 320 < 2) ? db5[i - 320] : 0.f;
    else if (i < 400) v = cb0[i - 336];
    else              v = (i - 400 < 3) ? cb1[i - 400] : 0.f;
    biasF[i] = v;
  }
  __syncthreads();

  unsigned short* act  = lds_us + ACT_OFF + wave * ACT_PER_WAVE;
  float* dispBuf       = (float*)(lds_us + DISP_OFF_US) + wave * 32;

  const float a0x = aabb[0], a0y = aabb[1], a0z = aabb[2];
  const float a1x = aabb[3], a1y = aabb[4], a1z = aabb[5];

  const int nTiles = nPoints >> 4;
  for (int tile = blockIdx.x * WAVES + wave; tile < nTiles; tile += gridDim.x * WAVES) {
    const int tileBase = tile * 16;
    int p = tileBase + m;
    if (p >= nPoints) p = nPoints - 1;

    // ---- normalize coords ----
    const float i0 = inp[p * 3 + 0], i1 = inp[p * 3 + 1], i2 = inp[p * 3 + 2];
    const float nx = (i0 - a0x) / (a1x - a0x) * 2.f - 1.f;
    const float ny = (i1 - a0y) / (a1y - a0y) * 2.f - 1.f;
    const float nz = (i2 - a0z) / (a1z - a0z) * 2.f - 1.f;

    // ---- triplane gathers -> feat bf16 rows, cols 0..191 ----
    sample_plane(plane01, 512, 512, nx, ny, act, m, half, 0);
    sample_plane(plane02, 512,  64, nx, nz, act, m, half, 64);
    sample_plane(plane12, 512,  64, ny, nz, act, m, half, 128);

    // ---- MLP decoder (h lives in cols 192..255; concat(feat,h) == cols 0..255) ----
    mlp_layer64<192>(act,   0, lds_us + W0_OFF, 192, biasF + B0_F, act, 192, m, half);
    mlp_layer64< 64>(act, 192, lds_us + W1_OFF,  64, biasF + B1_F, act, 192, m, half);
    mlp_layer64<256>(act,   0, lds_us + W2_OFF, 256, biasF + B2_F, act, 192, m, half);
    mlp_layer64< 64>(act, 192, lds_us + W3_OFF,  64, biasF + B3_F, act, 192, m, half);
    mlp_layer64<256>(act,   0, lds_us + W4_OFF, 256, biasF + B4_F, act, 192, m, half);

    // ---- disp = h @ W5^T + b5 (padded 16-wide, cols 0,1 valid) ----
    {
      v8f acc = gemm16(act, 192, 64, lds_us + W5_OFF, 64, m, half);
      const int rb = half * 8;
      if (m < 2) {
        const float bs = biasF[B5_F + m];
#pragma unroll
        for (int r = 0; r < 8; ++r) {
          const int row = rb + r;
          const float o = acc[r] + bs;
          dispBuf[row * 2 + m] = o;
          const int gp = tileBase + row;
          if (gp < nPoints) out[(size_t)gp * 2 + m] = o;
        }
      }
    }

    // ---- canonical gather at displaced coords -> cols 192..255 ----
    const float dx = dispBuf[m * 2 + 0];
    const float dy = dispBuf[m * 2 + 1];
    sample_plane(can_plane, 512, 512, nx + dx, ny + dy, act, m, half, 192);

    // ---- color head ----
    mlp_layer64<64>(act, 192, lds_us + C0_OFF, 64, biasF + CB0_F, act, 192, m, half);
    {
      v8f acc = gemm16(act, 192, 64, lds_us + C1_OFF, 64, m, half);
      const int rb = half * 8;
      if (m < 3) {
        const float bs = biasF[CB1_F + m];
        const size_t rgbBase = (size_t)nPoints * 2;
#pragma unroll
        for (int r = 0; r < 8; ++r) {
          const int gp = tileBase + rb + r;
          if (gp < nPoints) out[rgbBase + (size_t)gp * 3 + m] = acc[r] + bs;
        }
      }
    }
  }
}

// ---------------- host launch ----------------
extern "C" void kernel_launch(void* const* d_in, const int* in_sizes, int n_in,
                              void* d_out, int out_size, void* d_ws, size_t ws_size,
                              hipStream_t stream) {
  const float* inp      = (const float*)d_in[0];
  const float* aabb     = (const float*)d_in[1];
  const float* plane01  = (const float*)d_in[2];
  const float* plane02  = (const float*)d_in[3];
  const float* plane12  = (const float*)d_in[4];
  const float* can_pl   = (const float*)d_in[5];
  const float* dW0 = (const float*)d_in[6];  const float* db0 = (const float*)d_in[7];
  const float* dW1 = (const float*)d_in[8];  const float* db1 = (const float*)d_in[9];
  const float* dW2 = (const float*)d_in[10]; const float* db2 = (const float*)d_in[11];
  const float* dW3 = (const float*)d_in[12]; const float* db3 = (const float*)d_in[13];
  const float* dW4 = (const float*)d_in[14]; const float* db4 = (const float*)d_in[15];
  const float* dW5 = (const float*)d_in[16]; const float* db5 = (const float*)d_in[17];
  const float* cW0 = (const float*)d_in[18]; const float* cb0 = (const float*)d_in[19];
  const float* cW1 = (const float*)d_in[20]; const float* cb1 = (const float*)d_in[21];

  const int nPoints = in_sizes[0] / 3;
  const int nTiles  = (nPoints + 15) / 16;
  int blocks = (nTiles + WAVES - 1) / WAVES;
  if (blocks < 1) blocks = 1;

  (void)hipFuncSetAttribute((const void*)triplane_disp_kernel,
                            hipFuncAttributeMaxDynamicSharedMemorySize, SHMEM_BYTES);

  triplane_disp_kernel<<<blocks, BLOCK, SHMEM_BYTES, stream>>>(
      inp, aabb, plane01, plane02, plane12, can_pl,
      dW0, db0, dW1, db1, dW2, db2, dW3, db3, dW4, db4, dW5, db5,
      cW0, cb0, cW1, cb1, (float*)d_out, nPoints);
}
